// ConservativePotential_6717328850966
// MI455X (gfx1250) — matrix operand order repaired and checked by reading, not compiled, not run on hardware
//
#include <hip/hip_runtime.h>
#include <hip/hip_bf16.h>
#include <math.h>

// ---- problem constants (match reference) ----
static constexpr int N_ATOMS = 20000;
static constexpr int E_LOC   = 320000;
static constexpr int E_GLOB  = 640000;
static constexpr int NGRAPH  = 256;
static constexpr int NLAYER  = 5;

typedef __attribute__((ext_vector_type(16))) _Float16 v16h_t;
typedef __attribute__((ext_vector_type(8)))  _Float16 v8h_t;
typedef __attribute__((ext_vector_type(8)))  float    v8f_t;

__device__ __forceinline__ float silu_f(float x) { return x / (1.0f + __expf(-x)); }

// ---- clamp embedding table rows to max-norm 1 ----
__global__ __launch_bounds__(64) void clamp_table_kernel(const float* __restrict__ t,
                                                         float* __restrict__ o, int cols) {
    __shared__ float red[64];
    int r = blockIdx.x, c = threadIdx.x;
    float val = (c < cols) ? t[r * cols + c] : 0.0f;
    red[c] = val * val;
    __syncthreads();
    for (int st = 32; st > 0; st >>= 1) { if (c < st) red[c] += red[c + st]; __syncthreads(); }
    float nrm = sqrtf(red[0]);
    float sc = fminf(1.0f, 1.0f / fmaxf(nrm, 1e-12f));
    if (c < cols) o[r * cols + c] = val * sc;
}

// ---- convert We (L,32,96) to f16, transposed to [L][set][96][32] ----
__global__ void convert_we_kernel(const float* __restrict__ Wl, const float* __restrict__ Wg,
                                  _Float16* __restrict__ WeT) {
    int idx = blockIdx.x * blockDim.x + threadIdx.x;
    const int TOTAL = NLAYER * 2 * 96 * 32;
    if (idx >= TOTAL) return;
    int k = idx & 31;
    int rem = idx >> 5;          // (l*2+set)*96 + n
    int n = rem % 96;
    int ls = rem / 96;           // l*2+set
    int set = ls & 1, l = ls >> 1;
    const float* src = set ? Wg : Wl;
    WeT[idx] = (_Float16)src[(size_t)(l * 32 + k) * 96 + n];
}

// ---- time-embedding MLP: silu(silu(t@W1+b1)@W2+b2), per-graph ----
__global__ __launch_bounds__(64) void temb_kernel(const float* __restrict__ t,
                                                  const float* __restrict__ W1, const float* __restrict__ b1,
                                                  const float* __restrict__ W2, const float* __restrict__ b2,
                                                  float* __restrict__ out) {
    __shared__ float tin[64], h1[64];
    int g = blockIdx.x, c = threadIdx.x;
    tin[c] = t[(size_t)g * 64 + c];
    __syncthreads();
    float a = b1[c];
    for (int k = 0; k < 64; ++k) a += tin[k] * W1[k * 64 + c];
    h1[c] = silu_f(a);
    __syncthreads();
    float a2 = b2[c];
    for (int k = 0; k < 64; ++k) a2 += h1[k] * W2[k * 64 + c];
    out[(size_t)g * 64 + c] = silu_f(a2);
}

// ---- initial node scalars: s = atom_embed[x] + temb[batch] ----
__global__ void init_s_kernel(const int* __restrict__ x, const int* __restrict__ batch,
                              const float* __restrict__ atomc, const float* __restrict__ temb,
                              float* __restrict__ s) {
    int idx = blockIdx.x * blockDim.x + threadIdx.x;
    if (idx >= N_ATOMS * 64) return;
    int n = idx >> 6, c = idx & 63;
    s[idx] = atomc[x[n] * 64 + c] + temb[(size_t)batch[n] * 64 + c];
}

// ---- per-target degrees (into deg buffer, later inverted) ----
__global__ void deg_kernel(const int* __restrict__ ei, int E, float* __restrict__ deg) {
    int e = blockIdx.x * blockDim.x + threadIdx.x;
    if (e >= E) return;
    atomicAdd(&deg[ei[E + e]], 1.0f);
}
__global__ void inv_kernel(float* __restrict__ d, int n) {
    int i = blockIdx.x * blockDim.x + threadIdx.x;
    if (i >= n) return;
    d[i] = 1.0f / fmaxf(d[i], 1.0f);
}

// ---- edge geometry + RBF + bond embedding -> f16 features (E x 32) ----
__global__ void edge_feat_kernel(const float* __restrict__ pos, const int* __restrict__ ei,
                                 const int* __restrict__ attr, const float* __restrict__ bondc,
                                 int E, float* __restrict__ rn, _Float16* __restrict__ feat) {
    int e = blockIdx.x * blockDim.x + threadIdx.x;
    if (e >= E) return;
    int src = ei[e], tgt = ei[E + e];
    float rx = pos[(size_t)tgt * 3 + 0] - pos[(size_t)src * 3 + 0];
    float ry = pos[(size_t)tgt * 3 + 1] - pos[(size_t)src * 3 + 1];
    float rz = pos[(size_t)tgt * 3 + 2] - pos[(size_t)src * 3 + 2];
    float d2 = fmaxf(rx * rx + ry * ry + rz * rz, 1e-6f);
    float d = sqrtf(d2), invd = 1.0f / d;
    rn[(size_t)e * 3 + 0] = rx * invd;
    rn[(size_t)e * 3 + 1] = ry * invd;
    rn[(size_t)e * 3 + 2] = rz * invd;
    const float PI_F = 3.14159265358979323846f;
    float env = 0.5f * (__cosf(PI_F * fminf(d * 0.1f, 1.0f)) + 1.0f);
    const float gamma = 2.56f;                  // (16/10)^2
    const float step = 10.0f / 15.0f;           // linspace(0,10,16)
    _Float16* frow = feat + (size_t)e * 32;
    #pragma unroll
    for (int k = 0; k < 16; ++k) {
        float diff = d - (float)k * step;
        frow[k] = (_Float16)(__expf(-gamma * diff * diff) * env);
    }
    const float* brow = bondc + (size_t)attr[e] * 16;
    #pragma unroll
    for (int k = 0; k < 16; ++k) frow[16 + k] = (_Float16)brow[k];
}

// ---- fused edge MLP (WMMA f16->f32) + message scatter (segment-mean via inv_deg) ----
// One wave handles 16 edges: A = feat tile (16x32 f16), B = WeT tile (32x16 f16),
// 6 N-tiles cover 96 outputs = [ws(64) | wv1(16) | wv2(16)].
__global__ __launch_bounds__(256) void edge_msg_kernel(
    const _Float16* __restrict__ feat, const float* __restrict__ rn,
    const int* __restrict__ ei, int E,
    const float* __restrict__ inv_deg,
    const _Float16* __restrict__ WeT,   // [96][32] f16 for this (layer,set)
    const float* __restrict__ be,       // [96]
    const float* __restrict__ s, const float* __restrict__ v,
    float* __restrict__ agg_s, float* __restrict__ agg_v) {
    const int lane = threadIdx.x & 31;
    const int wid  = blockIdx.x * (blockDim.x >> 5) + (threadIdx.x >> 5);
    const int e0   = wid << 4;            // 16 edges per wave (E divisible by 128/block)
    const int col  = lane & 15;
    const int hi   = lane >> 4;           // 0: K 0-15 & rows 0-7 ; 1: K 16-31 & rows 8-15
    const int kb   = hi << 3;

    // A: contiguous K order (k = 16*hi + s), matching B
    const _Float16* arow = feat + (size_t)(e0 + col) * 32;
    union { v16h_t v16; v8h_t v8[2]; } au;
    au.v16 = *(const v16h_t*)(arow + hi * 16);
    v16h_t a = au.v16;

    v8f_t acc[6];
    #pragma unroll
    for (int t = 0; t < 6; ++t) {
        int cb = t * 16 + col;
        v16h_t b = *(const v16h_t*)(WeT + (size_t)cb * 32 + hi * 16);
        float bias = be[cb];
        v8f_t c = { bias, bias, bias, bias, bias, bias, bias, bias };
        acc[t] = __builtin_amdgcn_wmma_f32_16x16x32_f16(false, a, false, b, (short)0, c,
                                                        false, false);
    }

    // D layout: VGPR r -> row (hi*8 + r), column = col. Scatter messages.
    const int rb = hi << 3;
    #pragma unroll
    for (int r = 0; r < 8; ++r) {
        int e   = e0 + rb + r;
        int src = ei[e];
        int tgt = ei[E + e];
        float inv = inv_deg[tgt];
        const float* ssrc = s + (size_t)src * 64;
        float* as = agg_s + (size_t)tgt * 64;
        #pragma unroll
        for (int t = 0; t < 4; ++t) {
            int ch = t * 16 + col;
            atomicAdd(&as[ch], acc[t][r] * ssrc[ch] * inv);
        }
        float wv1 = acc[4][r], wv2 = acc[5][r];
        const float* vsrc = v + (size_t)src * 48;
        float* av = agg_v + (size_t)tgt * 48;
        #pragma unroll
        for (int ax = 0; ax < 3; ++ax) {
            float mv = wv1 * vsrc[ax * 16 + col] + wv2 * rn[(size_t)e * 3 + ax];
            atomicAdd(&av[ax * 16 + col], mv * inv);
        }
    }
}

// ---- node update: MLP + residual + layernorm, and v += agg_v ----
__global__ __launch_bounds__(64) void node_update_kernel(
    float* __restrict__ s, const float* __restrict__ agg_s,
    const float* __restrict__ agg_v, float* __restrict__ v,
    const float* __restrict__ Wu1, const float* __restrict__ bu1,
    const float* __restrict__ Wu2, const float* __restrict__ bu2,
    const float* __restrict__ lng, const float* __restrict__ lnb) {
    __shared__ float cat[128];
    __shared__ float hb[64];
    __shared__ float r1[64], r2[64];
    int n = blockIdx.x, c = threadIdx.x;
    float sc = s[(size_t)n * 64 + c];
    cat[c]      = sc;
    cat[64 + c] = agg_s[(size_t)n * 64 + c];
    __syncthreads();
    float a1 = bu1[c];
    for (int k = 0; k < 128; ++k) a1 += cat[k] * Wu1[k * 64 + c];
    hb[c] = silu_f(a1);
    __syncthreads();
    float a2 = bu2[c];
    for (int k = 0; k < 64; ++k) a2 += hb[k] * Wu2[k * 64 + c];
    float h = sc + a2;
    r1[c] = h; r2[c] = h * h;
    __syncthreads();
    for (int st = 32; st > 0; st >>= 1) {
        if (c < st) { r1[c] += r1[c + st]; r2[c] += r2[c + st]; }
        __syncthreads();
    }
    float mu  = r1[0] * (1.0f / 64.0f);
    float var = r2[0] * (1.0f / 64.0f) - mu * mu;
    s[(size_t)n * 64 + c] = (h - mu) * rsqrtf(var + 1e-5f) * lng[c] + lnb[c];
    if (c < 48) v[(size_t)n * 48 + c] += agg_v[(size_t)n * 48 + c];
}

// ---- readout: vector norms + MLP head + per-graph segment sum ----
__global__ __launch_bounds__(64) void readout_kernel(
    const float* __restrict__ s, const float* __restrict__ v,
    const float* __restrict__ Wvd, const float* __restrict__ Wd1,
    const float* __restrict__ bd1, const float* __restrict__ Wd2,
    const float* __restrict__ bd2, const int* __restrict__ batch,
    float* __restrict__ out) {
    __shared__ float cat[80];
    __shared__ float red[64];
    int n = blockIdx.x, c = threadIdx.x;
    if (c < 16) {
        float vp0 = 0.f, vp1 = 0.f, vp2 = 0.f;
        const float* vn = v + (size_t)n * 48;
        for (int k = 0; k < 16; ++k) {
            float w = Wvd[k * 16 + c];
            vp0 += vn[k] * w;
            vp1 += vn[16 + k] * w;
            vp2 += vn[32 + k] * w;
        }
        cat[64 + c] = sqrtf(vp0 * vp0 + vp1 * vp1 + vp2 * vp2 + 1e-8f);
    }
    cat[c] = s[(size_t)n * 64 + c];
    __syncthreads();
    float a = bd1[c];
    for (int k = 0; k < 80; ++k) a += cat[k] * Wd1[k * 64 + c];
    red[c] = silu_f(a) * Wd2[c];
    __syncthreads();
    for (int st = 32; st > 0; st >>= 1) { if (c < st) red[c] += red[c + st]; __syncthreads(); }
    if (c == 0) atomicAdd(&out[batch[n]], red[0] + bd2[0]);
}

static inline size_t align256(size_t x) { return (x + 255) & ~(size_t)255; }

extern "C" void kernel_launch(void* const* d_in, const int* in_sizes, int n_in,
                              void* d_out, int out_size, void* d_ws, size_t ws_size,
                              hipStream_t stream) {
    (void)in_sizes; (void)n_in; (void)ws_size;
    const int*   x        = (const int*)d_in[0];
    const float* t        = (const float*)d_in[1];
    const float* pos      = (const float*)d_in[2];
    const int*   eil      = (const int*)d_in[3];
    const int*   eig      = (const int*)d_in[4];
    const int*   eal      = (const int*)d_in[5];
    const int*   eag      = (const int*)d_in[6];
    const int*   batch    = (const int*)d_in[7];
    const float* atom_tab = (const float*)d_in[8];
    const float* bond_tab = (const float*)d_in[9];
    const float* tW1      = (const float*)d_in[10];
    const float* tb1      = (const float*)d_in[11];
    const float* tW2      = (const float*)d_in[12];
    const float* tb2      = (const float*)d_in[13];
    const float* We_loc   = (const float*)d_in[14];
    const float* be_loc   = (const float*)d_in[15];
    const float* We_glob  = (const float*)d_in[16];
    const float* be_glob  = (const float*)d_in[17];
    const float* Wu1      = (const float*)d_in[18];
    const float* bu1      = (const float*)d_in[19];
    const float* Wu2      = (const float*)d_in[20];
    const float* bu2      = (const float*)d_in[21];
    const float* ln_g     = (const float*)d_in[22];
    const float* ln_b     = (const float*)d_in[23];
    const float* Wv_down  = (const float*)d_in[24];
    const float* Wd1      = (const float*)d_in[25];
    const float* bd1      = (const float*)d_in[26];
    const float* Wd2      = (const float*)d_in[27];
    const float* bd2      = (const float*)d_in[28];

    // ---- carve workspace ----
    char* p = (char*)d_ws;
    auto carve = [&](size_t bytes) -> char* { char* r = p; p += align256(bytes); return r; };
    float*    s      = (float*)carve((size_t)N_ATOMS * 64 * 4);
    float*    v      = (float*)carve((size_t)N_ATOMS * 48 * 4);
    float*    agg    = (float*)carve((size_t)N_ATOMS * 112 * 4);  // agg_s | agg_v contiguous
    float*    agg_s  = agg;
    float*    agg_v  = agg + (size_t)N_ATOMS * 64;
    float*    tembuf = (float*)carve((size_t)NGRAPH * 64 * 4);
    float*    atomc  = (float*)carve((size_t)100 * 64 * 4);
    float*    bondc  = (float*)carve((size_t)5 * 16 * 4);
    float*    inv_l  = (float*)carve((size_t)N_ATOMS * 4);
    float*    inv_g  = (float*)carve((size_t)N_ATOMS * 4);
    float*    rn_l   = (float*)carve((size_t)E_LOC * 3 * 4);
    float*    rn_g   = (float*)carve((size_t)E_GLOB * 3 * 4);
    _Float16* feat_l = (_Float16*)carve((size_t)E_LOC * 32 * 2);
    _Float16* feat_g = (_Float16*)carve((size_t)E_GLOB * 32 * 2);
    _Float16* WeT    = (_Float16*)carve((size_t)NLAYER * 2 * 96 * 32 * 2);

    // ---- preprocessing ----
    hipMemsetAsync(inv_l, 0, (size_t)N_ATOMS * 4, stream);
    hipMemsetAsync(inv_g, 0, (size_t)N_ATOMS * 4, stream);
    hipMemsetAsync(v, 0, (size_t)N_ATOMS * 48 * 4, stream);
    hipMemsetAsync(d_out, 0, (size_t)out_size * 4, stream);

    clamp_table_kernel<<<100, 64, 0, stream>>>(atom_tab, atomc, 64);
    clamp_table_kernel<<<5, 64, 0, stream>>>(bond_tab, bondc, 16);
    convert_we_kernel<<<(NLAYER * 2 * 96 * 32 + 255) / 256, 256, 0, stream>>>(We_loc, We_glob, WeT);
    temb_kernel<<<NGRAPH, 64, 0, stream>>>(t, tW1, tb1, tW2, tb2, tembuf);
    init_s_kernel<<<(N_ATOMS * 64 + 255) / 256, 256, 0, stream>>>(x, batch, atomc, tembuf, s);
    deg_kernel<<<(E_LOC + 255) / 256, 256, 0, stream>>>(eil, E_LOC, inv_l);
    deg_kernel<<<(E_GLOB + 255) / 256, 256, 0, stream>>>(eig, E_GLOB, inv_g);
    inv_kernel<<<(N_ATOMS + 255) / 256, 256, 0, stream>>>(inv_l, N_ATOMS);
    inv_kernel<<<(N_ATOMS + 255) / 256, 256, 0, stream>>>(inv_g, N_ATOMS);
    edge_feat_kernel<<<(E_LOC + 255) / 256, 256, 0, stream>>>(pos, eil, eal, bondc, E_LOC, rn_l, feat_l);
    edge_feat_kernel<<<(E_GLOB + 255) / 256, 256, 0, stream>>>(pos, eig, eag, bondc, E_GLOB, rn_g, feat_g);

    // ---- message-passing layers ----
    for (int l = 0; l < NLAYER; ++l) {
        hipMemsetAsync(agg, 0, (size_t)N_ATOMS * 112 * 4, stream);
        edge_msg_kernel<<<E_LOC / 128, 256, 0, stream>>>(
            feat_l, rn_l, eil, E_LOC, inv_l,
            WeT + (size_t)(l * 2 + 0) * 96 * 32, be_loc + (size_t)l * 96,
            s, v, agg_s, agg_v);
        edge_msg_kernel<<<E_GLOB / 128, 256, 0, stream>>>(
            feat_g, rn_g, eig, E_GLOB, inv_g,
            WeT + (size_t)(l * 2 + 1) * 96 * 32, be_glob + (size_t)l * 96,
            s, v, agg_s, agg_v);
        node_update_kernel<<<N_ATOMS, 64, 0, stream>>>(
            s, agg_s, agg_v, v,
            Wu1 + (size_t)l * 128 * 64, bu1 + (size_t)l * 64,
            Wu2 + (size_t)l * 64 * 64, bu2 + (size_t)l * 64,
            ln_g + (size_t)l * 64, ln_b + (size_t)l * 64);
    }

    // ---- readout ----
    readout_kernel<<<N_ATOMS, 64, 0, stream>>>(s, v, Wv_down, Wd1, bd1, Wd2, bd2, batch,
                                               (float*)d_out);
}
